// SiameseNetwork_57896159150736
// MI455X (gfx1250) — matrix operand order repaired and checked by reading, not compiled, not run on hardware
//
#include <hip/hip_runtime.h>

// ---------- types ----------
typedef __attribute__((ext_vector_type(16))) __bf16 v16bf;
typedef __attribute__((ext_vector_type(8)))  float  v8f;
struct alignas(16) U4 { unsigned x, y, z, w; };

__device__ __forceinline__ __bf16 f2bf(float f) {
    union { float f; unsigned u; } in; in.f = f;
    unsigned u = in.u;
    unsigned lsb = (u >> 16) & 1u;
    u += 0x7FFFu + lsb;                       // round-to-nearest-even
    union { unsigned short s; __bf16 b; } out;
    out.s = (unsigned short)(u >> 16);
    return out.b;
}

#define DIMS 512
#define NNEI 31
#define NOUT 300

// ---------- attention head: gather + scores + softmax + context -> bf16 feat[b,1024] ----------
__global__ __launch_bounds__(256) void attend_kernel(
    const float* __restrict__ emb,
    const int*   __restrict__ idx,    // base pointer for this head
    int strideB,                       // ints per batch row
    const float* __restrict__ vvec,    // (31,)
    __bf16* __restrict__ feat)         // [B][1024] row-major
{
    __shared__ float s_neigh[NNEI * DIMS];   // 63,488 B
    __shared__ float s_w[NNEI + 1];

    const int b    = blockIdx.x;
    const int t    = threadIdx.x;
    const int lane = t & 31;
    const int wave = t >> 5;

    const int* row = idx + (size_t)b * strideB;
    const int  i0  = row[0];
    const float* nodep = emb + (size_t)i0 * DIMS;
    __bf16* fb = feat + (size_t)b * 1024;

    // node -> feat[0:512] (bf16), plus strided copy into registers for the dots
    {
        float2 nv = *(const float2*)(nodep + t * 2);
        fb[t * 2]     = f2bf(nv.x);
        fb[t * 2 + 1] = f2bf(nv.y);
    }
    float nodereg[16];
    #pragma unroll
    for (int i = 0; i < 16; ++i) nodereg[i] = nodep[lane + 32 * i];

    // stage all 31 neighbor rows in LDS
    for (int j = 0; j < NNEI; ++j) {
        const float* np = emb + (size_t)row[1 + j] * DIMS;
        float2 nv = *(const float2*)(np + t * 2);
        s_neigh[j * DIMS + t * 2]     = nv.x;
        s_neigh[j * DIMS + t * 2 + 1] = nv.y;
    }
    __syncthreads();

    // scores: wave w handles neighbors w, w+8, w+16, w+24
    for (int j = wave; j < NNEI; j += 8) {
        float acc = 0.f;
        #pragma unroll
        for (int i = 0; i < 16; ++i)
            acc += s_neigh[j * DIMS + lane + 32 * i] * nodereg[i];
        #pragma unroll
        for (int off = 16; off > 0; off >>= 1)
            acc += __shfl_xor(acc, off, 32);
        if (lane == 0) s_w[j] = acc;
    }
    __syncthreads();

    // masked softmax * v  (tiny: single thread)
    if (t == 0) {
        float sc[NNEI];
        float m = -3.4e38f;
        for (int j = 0; j < NNEI; ++j) {
            float s = s_w[j];
            if (s == 0.0f) s -= 9999.0f;      // scores + where(s!=0,0,-1)*NEG
            sc[j] = s;
            m = fmaxf(m, s);
        }
        float sum = 0.f;
        for (int j = 0; j < NNEI; ++j) { sc[j] = __expf(sc[j] - m); sum += sc[j]; }
        float inv = 1.0f / sum;
        for (int j = 0; j < NNEI; ++j) s_w[j] = sc[j] * inv * vvec[j];
    }
    __syncthreads();

    // context -> feat[512:1024]
    float c0 = 0.f, c1 = 0.f;
    #pragma unroll 1
    for (int j = 0; j < NNEI; ++j) {
        float w = s_w[j];
        c0 += w * s_neigh[j * DIMS + t * 2];
        c1 += w * s_neigh[j * DIMS + t * 2 + 1];
    }
    fb[DIMS + t * 2]     = f2bf(c0);
    fb[DIMS + t * 2 + 1] = f2bf(c1);
}

// ---------- f32 -> bf16 weight convert with padding ----------
__global__ __launch_bounds__(256) void convert_w(
    const float* __restrict__ W, __bf16* __restrict__ Wb,
    int rows, int cols, int rowsPad, int colsPad)
{
    int gid = blockIdx.x * 256 + threadIdx.x;
    if (gid >= rowsPad * colsPad) return;
    int r = gid / colsPad, c = gid % colsPad;
    float val = (r < rows && c < cols) ? W[(size_t)r * cols + c] : 0.0f;
    Wb[gid] = f2bf(val);
}

// ---------- WMMA bf16 GEMM:  out[M,300] = A[M,K] * W[304,K]^T + bias ----------
__global__ __launch_bounds__(256) void gemm_wmma(
    const __bf16* __restrict__ A,     // M x K row-major bf16
    const __bf16* __restrict__ W,     // 304 x K row-major bf16 (rows >= 300 are zero)
    const float*  __restrict__ bias,  // (300,)
    float* __restrict__ out,          // M x 300 row-major f32
    int M, int N, int K, int tilesN)
{
    const int wid  = threadIdx.x >> 5;
    const int lane = threadIdx.x & 31;
    const int gw   = blockIdx.x * (blockDim.x >> 5) + wid;
    const int tN   = gw % tilesN;
    const int tM   = gw / tilesN;
    if (tM * 16 >= M) return;

    const int nlo  = lane & 15;
    const int aoff = (lane < 16) ? 0 : 8;    // A: 16-bit 16x32 lane layout
    const int boff = (lane < 16) ? 0 : 8;    // B: 32x16 lane layout

    const __bf16* arow = A + (size_t)(tM * 16 + nlo) * K;
    const __bf16* wrow = W + (size_t)(tN * 16 + nlo) * K;

    v8f acc = {};
    for (int k = 0; k < K; k += 32) {
        union { v16bf v; U4 u[2]; } a, bm;
        a.u[0]  = *(const U4*)(arow + k + aoff);
        a.u[1]  = *(const U4*)(arow + k + aoff + 16);
        bm.u[0] = *(const U4*)(wrow + k + boff);
        bm.u[1] = *(const U4*)(wrow + k + boff + 16);
        __builtin_prefetch((const void*)(arow + k + 128), 0, 0);
        acc = __builtin_amdgcn_wmma_f32_16x16x32_bf16(
                  false, a.v, false, bm.v, (short)0, acc, false, false);
    }

    const int mo = tM * 16 + ((lane < 16) ? 0 : 8);
    const int n  = tN * 16 + nlo;
    if (n < N) {
        float bv = bias[n];
        #pragma unroll
        for (int r = 0; r < 8; ++r)
            out[(size_t)(mo + r) * N + n] = acc[r] + bv;
    }
}

// ---------- assemble prop feature: [dom(300) | rng(300) | prp(512) | 0-pad(8)] -> bf16 ----------
__global__ __launch_bounds__(256) void assemble_prop(
    const float* __restrict__ dom,
    const float* __restrict__ rng,
    const float* __restrict__ emb,
    const int*   __restrict__ idxProp,  // base at (:, i); stride 192 per b; element [b*192+0] = prp idx
    __bf16* __restrict__ featP,         // [B][1120]
    int B)
{
    int gid = blockIdx.x * 256 + threadIdx.x;
    if (gid >= B * 1120) return;
    int b = gid / 1120, c = gid % 1120;
    float val;
    if (c < 300)       val = dom[(size_t)b * NOUT + c];
    else if (c < 600)  val = rng[(size_t)b * NOUT + (c - 300)];
    else if (c < 1112) {
        int pi = idxProp[(size_t)b * 192];
        val = emb[(size_t)pi * DIMS + (c - 600)];
    } else val = 0.0f;
    featP[gid] = f2bf(val);
}

// ---------- cosine: blocks [0,B) -> ent, [B,2B) -> prop ----------
__global__ __launch_bounds__(32) void cosine_kernel(
    const float* __restrict__ r0, const float* __restrict__ r1,
    const float* __restrict__ p0, const float* __restrict__ p1,
    float* __restrict__ out, int B)
{
    int b = blockIdx.x, lane = threadIdx.x;
    const float *A, *C; int rrow;
    if (b < B) { A = r0; C = r1; rrow = b; }
    else       { A = p0; C = p1; rrow = b - B; }
    float d = 0.f, na = 0.f, nb = 0.f;
    for (int k = lane; k < NOUT; k += 32) {
        float x = A[(size_t)rrow * NOUT + k];
        float y = C[(size_t)rrow * NOUT + k];
        d += x * y; na += x * x; nb += y * y;
    }
    #pragma unroll
    for (int off = 16; off > 0; off >>= 1) {
        d  += __shfl_xor(d,  off, 32);
        na += __shfl_xor(na, off, 32);
        nb += __shfl_xor(nb, off, 32);
    }
    if (lane == 0)
        out[b] = d / (fmaxf(sqrtf(na), 1e-8f) * fmaxf(sqrtf(nb), 1e-8f));
}

// ---------- orchestration ----------
extern "C" void kernel_launch(void* const* d_in, const int* in_sizes, int n_in,
                              void* d_out, int out_size, void* d_ws, size_t ws_size,
                              hipStream_t stream) {
    const int*   inputs      = (const int*)  d_in[0];  // (B,2,32)
    const int*   inputs_prop = (const int*)  d_in[1];  // (B,2,3,32)
    const float* emb         = (const float*)d_in[2];  // (V,512)
    const float* W_out       = (const float*)d_in[3];  // (300,1024)
    const float* b_out       = (const float*)d_in[4];  // (300,)
    const float* v           = (const float*)d_in[5];  // (31,)
    const float* W_prop      = (const float*)d_in[6];  // (300,1112)
    const float* b_prop      = (const float*)d_in[7];  // (300,)
    float* out = (float*)d_out;

    const int B = in_sizes[0] / 64;          // 4096
    const int N = NOUT, K1 = 1024, K2 = 1120, NPAD = 304;
    const int TILES_N = (N + 15) / 16;       // 19

    char* ws = (char*)d_ws;
    size_t off = 0;
    auto carve = [&](size_t bytes) -> char* {
        char* p = ws + off;
        off += (bytes + 255) & ~(size_t)255;
        return p;
    };
    __bf16* feat   = (__bf16*)carve((size_t)B * 1120 * 2);
    __bf16* woutb  = (__bf16*)carve((size_t)NPAD * K1 * 2);
    __bf16* wpropb = (__bf16*)carve((size_t)NPAD * K2 * 2);
    float*  res0   = (float*) carve((size_t)B * N * 4);
    float*  res1   = (float*) carve((size_t)B * N * 4);
    float*  dom    = (float*) carve((size_t)B * N * 4);
    float*  rngv   = (float*) carve((size_t)B * N * 4);
    float*  prop0  = (float*) carve((size_t)B * N * 4);
    float*  prop1  = (float*) carve((size_t)B * N * 4);

    // weight conversion (f32 -> padded bf16)
    {
        int tot1 = NPAD * K1;
        convert_w<<<(tot1 + 255) / 256, 256, 0, stream>>>(W_out, woutb, 300, 1024, NPAD, K1);
        int tot2 = NPAD * K2;
        convert_w<<<(tot2 + 255) / 256, 256, 0, stream>>>(W_prop, wpropb, 300, 1112, NPAD, K2);
    }

    const int gemmBlocks = ((B / 16) * TILES_N + 7) / 8;   // 8 waves / block

    // entity heads
    for (int i = 0; i < 2; ++i) {
        attend_kernel<<<B, 256, 0, stream>>>(emb, inputs + i * 32, 64, v, feat);
        gemm_wmma<<<gemmBlocks, 256, 0, stream>>>(feat, woutb, b_out,
                                                  i ? res1 : res0, B, N, K1, TILES_N);
    }

    // prop heads
    for (int i = 0; i < 2; ++i) {
        attend_kernel<<<B, 256, 0, stream>>>(emb, inputs_prop + i * 96 + 32, 192, v, feat);
        gemm_wmma<<<gemmBlocks, 256, 0, stream>>>(feat, woutb, b_out, dom, B, N, K1, TILES_N);

        attend_kernel<<<B, 256, 0, stream>>>(emb, inputs_prop + i * 96 + 64, 192, v, feat);
        gemm_wmma<<<gemmBlocks, 256, 0, stream>>>(feat, woutb, b_out, rngv, B, N, K1, TILES_N);

        assemble_prop<<<((B * 1120) + 255) / 256, 256, 0, stream>>>(
            dom, rngv, emb, inputs_prop + i * 96, feat, B);
        gemm_wmma<<<gemmBlocks, 256, 0, stream>>>(feat, wpropb, b_prop,
                                                  i ? prop1 : prop0, B, N, K2, TILES_N);
    }

    cosine_kernel<<<2 * B, 32, 0, stream>>>(res0, res1, prop0, prop1, out, B);
}
